// QRLoRALayer_18966575579817
// MI455X (gfx1250) — matrix operand order repaired and checked by reading, not compiled, not run on hardware
//
#include <hip/hip_runtime.h>
#include <stdint.h>

// ---------------------------------------------------------------------------
// QR-LoRA forward on MI455X (gfx1250, wave32, WMMA bf16).
//   Stage 1: W_eff = W + SCALING * Q @ diag(lam) @ R   (bf16, into d_ws)
//   Stage 2: out   = x @ W_eff.T                       (f32 out, bf16 WMMA)
// ---------------------------------------------------------------------------

typedef __attribute__((ext_vector_type(16))) __bf16 v16bf;
typedef __attribute__((ext_vector_type(8)))  __bf16 v8bf;
typedef __attribute__((ext_vector_type(4)))  __bf16 v4bf;
typedef __attribute__((ext_vector_type(2)))  __bf16 v2bf;
typedef __attribute__((ext_vector_type(8)))  float  v8f;
typedef __attribute__((ext_vector_type(4)))  int    vi4;

#define T_DIM   16384
#define IN_DIM  2048
#define OUT_DIM 2048
#define RANK    512
#define SCALING 2.0f

#define BM 128
#define BN 128
#define BK 64
#define LSTR 72            // LDS row stride in bf16 (64 data + 8 pad = 144 B)
#define NTHREADS 256

// -------------------- f32 -> bf16 conversion helpers -----------------------
#if __has_builtin(__builtin_amdgcn_cvt_pk_bf16_f32)
__device__ __forceinline__ v2bf cvt_pk(float a, float b) {
  return __builtin_amdgcn_cvt_pk_bf16_f32(a, b);   // single VALU op
}
#else
// Let the backend legalize fptrunc f32->bf16; on gfx1250 this can select the
// hardware V_CVT_PK_BF16_F32 instead of a 4-op software rounding sequence.
__device__ __forceinline__ v2bf cvt_pk(float a, float b) {
  v2bf r; r[0] = (__bf16)a; r[1] = (__bf16)b; return r;
}
#endif

__device__ __forceinline__ __bf16 f2bf(float f) { return cvt_pk(f, f)[0]; }

__device__ __forceinline__ v4bf cvt4(float4 f) {
  v2bf lo = cvt_pk(f.x, f.y), hi = cvt_pk(f.z, f.w);
  return __builtin_shufflevector(lo, hi, 0, 1, 2, 3);
}

// -------------------- async global->LDS copy (gfx1250) ---------------------
#if __has_builtin(__builtin_amdgcn_global_load_async_to_lds_b128) && \
    __has_builtin(__builtin_amdgcn_s_wait_asynccnt)
#define HAVE_ASYNC_LDS 1
// NOTE: the builtin wants a NON-const addrspace(1) (i.e. __device__) src ptr.
typedef __attribute__((address_space(1))) vi4 glb_vi4;
typedef __attribute__((address_space(3))) vi4 lds_vi4;
#else
#define HAVE_ASYNC_LDS 0
#endif

// -------------------- WMMA fragment helpers --------------------------------
// Assemble a 16-element bf16 WMMA fragment from two contiguous 16B LDS chunks.
__device__ __forceinline__ v16bf ld_frag(const __bf16* p, int off2_bytes) {
  v8bf lo = *(const v8bf*)p;
  v8bf hi = *(const v8bf*)((const char*)p + off2_bytes);
  return __builtin_shufflevector(lo, hi, 0,1,2,3,4,5,6,7,8,9,10,11,12,13,14,15);
}

// One BK=64 tile of WMMA work for one wave: 2 K-steps x (4x2) 16x16x32 MMAs.
// A fragment lane layout: row = lane%16, K in [16h,16h+16).
// B fragment lane layout: col = lane%16, K in [16h,16h+16).
__device__ __forceinline__ void mma_tile(const __bf16* bufA, const __bf16* bufB,
                                         int wm, int wn, int lane16, int h,
                                         v8f acc[4][2]) {
#pragma unroll
  for (int ks = 0; ks < 2; ++ks) {
    v16bf a[4], b[2];
#pragma unroll
    for (int mi = 0; mi < 4; ++mi)
      a[mi] = ld_frag(bufA + (wm*64 + mi*16 + lane16)*LSTR + ks*32 + 16*h, 16);
#pragma unroll
    for (int ni = 0; ni < 2; ++ni)
      b[ni] = ld_frag(bufB + (wn*32 + ni*16 + lane16)*LSTR + ks*32 + 16*h, 16);
#pragma unroll
    for (int mi = 0; mi < 4; ++mi)
#pragma unroll
      for (int ni = 0; ni < 2; ++ni)
        acc[mi][ni] = __builtin_amdgcn_wmma_f32_16x16x32_bf16(
            false, a[mi], false, b[ni], (short)0, acc[mi][ni], false, false);
  }
}

// ---------------------------------------------------------------------------
// Stage 1: W_eff[o][i] = W[o][i] + sum_r (s*lam[r]*Q[o][r]) * R[r][i]
// GEMM: M=OUT (o), N=IN (i), K=RANK (r). s*lam folded into A conversion.
// ---------------------------------------------------------------------------
__global__ __launch_bounds__(NTHREADS)
void build_weff_kernel(const float* __restrict__ W, const float* __restrict__ Q,
                       const float* __restrict__ R, const float* __restrict__ lam,
                       __bf16* __restrict__ weff) {
  __shared__ __bf16 sA[2][BM * LSTR];
  __shared__ __bf16 sB[2][BN * LSTR];

  const int t = threadIdx.x;
  const int lane = t & 31, wave = t >> 5;
  const int lane16 = lane & 15, h = lane >> 4;
  const int wm = wave >> 2, wn = wave & 3;          // 2x4 wave grid
  const int m0 = blockIdx.x * BM;                   // o tile
  const int n0 = blockIdx.y * BN;                   // i tile
  const int NT = RANK / BK;                         // 8 K-tiles

  v8f acc[4][2];
  const v8f vzero = {0.f,0.f,0.f,0.f,0.f,0.f,0.f,0.f};
#pragma unroll
  for (int mi = 0; mi < 4; ++mi)
#pragma unroll
    for (int ni = 0; ni < 2; ++ni) acc[mi][ni] = vzero;

  const int arow0 = t >> 4;            // A: rows arow0 + 16*l, col fixed
  const int acol  = (t & 15) << 2;     // A: 4 contiguous r per float4
  const int bi    = lane << 2;         // B: 4 contiguous i per float4
  float4 ra[8], rb[8];
  float4 sl;                           // s*lam[acol..acol+3] for current tile

  auto loadA = [&](int kt) {
    float4 l4 = *(const float4*)(lam + kt*BK + acol);
    sl.x = SCALING*l4.x; sl.y = SCALING*l4.y; sl.z = SCALING*l4.z; sl.w = SCALING*l4.w;
#pragma unroll
    for (int l = 0; l < 8; ++l)
      ra[l] = *(const float4*)(Q + (size_t)(m0 + arow0 + 16*l)*RANK + kt*BK + acol);
  };
  auto loadB = [&](int kt) {
#pragma unroll
    for (int l = 0; l < 8; ++l)
      rb[l] = *(const float4*)(R + (size_t)(kt*BK + wave + 8*l)*IN_DIM + n0 + bi);
  };
  auto storeA = [&](int buf) {
#pragma unroll
    for (int l = 0; l < 8; ++l) {
      float4 v = {ra[l].x*sl.x, ra[l].y*sl.y, ra[l].z*sl.z, ra[l].w*sl.w};
      *(v4bf*)&sA[buf][(arow0 + 16*l)*LSTR + acol] = cvt4(v);
    }
  };
  auto storeB = [&](int buf) {                      // transpose R tile -> [i][r]
#pragma unroll
    for (int l = 0; l < 8; ++l) {
      const int r = wave + 8*l;
      sB[buf][(bi + 0)*LSTR + r] = f2bf(rb[l].x);
      sB[buf][(bi + 1)*LSTR + r] = f2bf(rb[l].y);
      sB[buf][(bi + 2)*LSTR + r] = f2bf(rb[l].z);
      sB[buf][(bi + 3)*LSTR + r] = f2bf(rb[l].w);
    }
  };

  loadA(0); loadB(0);
  storeA(0); storeB(0);
  __syncthreads();
  for (int kt = 0; kt < NT; ++kt) {
    const int cur = kt & 1;
    if (kt + 1 < NT) { loadA(kt + 1); loadB(kt + 1); }
    mma_tile(sA[cur], sB[cur], wm, wn, lane16, h, acc);
    __syncthreads();
    if (kt + 1 < NT) {
      storeA(1 - cur); storeB(1 - cur);
      __syncthreads();
    }
  }

  // Epilogue: add W, convert to bf16, store W_eff row-major [OUT][IN].
#pragma unroll
  for (int mi = 0; mi < 4; ++mi) {
    const int o_base = m0 + wm*64 + mi*16 + 8*h;
#pragma unroll
    for (int ni = 0; ni < 2; ++ni) {
      const int icol = n0 + wn*32 + ni*16 + lane16;
#pragma unroll
      for (int v = 0; v < 8; ++v) {
        const size_t idx = (size_t)(o_base + v)*IN_DIM + icol;
        weff[idx] = f2bf(W[idx] + acc[mi][ni][v]);
      }
    }
  }
}

// ---------------------------------------------------------------------------
// Stage 2: out = x @ W_eff.T  (M=T, N=OUT, K=IN), x converted to bf16 on load.
// B tile (already bf16) is moved with gfx1250 async global->LDS copies.
// ---------------------------------------------------------------------------
__global__ __launch_bounds__(NTHREADS)
void qrlora_gemm_kernel(const float* __restrict__ x,
                        const __bf16* __restrict__ weff,
                        float* __restrict__ out) {
  __shared__ __bf16 sA[2][BM * LSTR];
  __shared__ __bf16 sB[2][BN * LSTR];

  const int t = threadIdx.x;
  const int lane = t & 31, wave = t >> 5;
  const int lane16 = lane & 15, h = lane >> 4;
  const int wm = wave >> 2, wn = wave & 3;
  const int m0 = blockIdx.x * BM;                   // token tile
  const int n0 = blockIdx.y * BN;                   // out-feature tile
  const int NT = IN_DIM / BK;                       // 32 K-tiles

  v8f acc[4][2];
  const v8f vzero = {0.f,0.f,0.f,0.f,0.f,0.f,0.f,0.f};
#pragma unroll
  for (int mi = 0; mi < 4; ++mi)
#pragma unroll
    for (int ni = 0; ni < 2; ++ni) acc[mi][ni] = vzero;

  const int arow0 = t >> 4;            // x rows arow0 + 16*l
  const int acol  = (t & 15) << 2;     // 4 contiguous k per float4
  const int brow0 = t >> 3;            // weff rows brow0 + 32*l (n index)
  const int bkc   = (t & 7) << 3;      // 8 contiguous bf16 per 16B chunk
  float4 ra[8];

  auto loadA = [&](int kt) {
#pragma unroll
    for (int l = 0; l < 8; ++l)
      ra[l] = *(const float4*)(x + (size_t)(m0 + arow0 + 16*l)*IN_DIM + kt*BK + acol);
    if (kt + 2 < NT)   // warm L2 ahead of the register prefetch
      __builtin_prefetch(x + (size_t)(m0 + arow0)*IN_DIM + (kt + 2)*BK + acol, 0, 1);
  };
  auto storeA = [&](int buf) {
#pragma unroll
    for (int l = 0; l < 8; ++l)
      *(v4bf*)&sA[buf][(arow0 + 16*l)*LSTR + acol] = cvt4(ra[l]);
  };

#if HAVE_ASYNC_LDS
  // bf16 B tile: straight async DMA global -> LDS, tracked by ASYNCcnt.
  auto copyB = [&](int kt, int buf) {
#pragma unroll
    for (int l = 0; l < 4; ++l) {
      glb_vi4* src = (glb_vi4*)(weff + (size_t)(n0 + brow0 + 32*l)*IN_DIM + kt*BK + bkc);
      lds_vi4* dst = (lds_vi4*)&sB[buf][(brow0 + 32*l)*LSTR + bkc];
      __builtin_amdgcn_global_load_async_to_lds_b128(src, dst, 0, 0);
    }
  };
#else
  uint4 rb[4];
  auto loadB = [&](int kt) {
#pragma unroll
    for (int l = 0; l < 4; ++l)
      rb[l] = *(const uint4*)(weff + (size_t)(n0 + brow0 + 32*l)*IN_DIM + kt*BK + bkc);
  };
  auto storeB = [&](int buf) {
#pragma unroll
    for (int l = 0; l < 4; ++l)
      *(uint4*)&sB[buf][(brow0 + 32*l)*LSTR + bkc] = rb[l];
  };
#endif

  // Prologue: tile 0 into buffer 0.
  loadA(0);
#if HAVE_ASYNC_LDS
  copyB(0, 0);
  storeA(0);
  __builtin_amdgcn_s_wait_asynccnt(0);
#else
  loadB(0);
  storeA(0); storeB(0);
#endif
  __syncthreads();

  for (int kt = 0; kt < NT; ++kt) {
    const int cur = kt & 1;
    if (kt + 1 < NT) loadA(kt + 1);
#if !HAVE_ASYNC_LDS
    if (kt + 1 < NT) loadB(kt + 1);
#endif
    mma_tile(sA[cur], sB[cur], wm, wn, lane16, h, acc);
    __syncthreads();                 // all waves done reading buf[1-cur]
    if (kt + 1 < NT) {
#if HAVE_ASYNC_LDS
      copyB(kt + 1, 1 - cur);        // DMA overlaps the A-tile conversion
      storeA(1 - cur);
      __builtin_amdgcn_s_wait_asynccnt(0);
#else
      storeA(1 - cur); storeB(1 - cur);
#endif
      __syncthreads();               // copies visible to all waves
    }
  }

  // Epilogue: f32 accumulators straight to out (row-major [T][OUT]).
#pragma unroll
  for (int mi = 0; mi < 4; ++mi) {
    const int trow = m0 + wm*64 + mi*16 + 8*h;
#pragma unroll
    for (int ni = 0; ni < 2; ++ni) {
      const int ocol = n0 + wn*32 + ni*16 + lane16;
#pragma unroll
      for (int v = 0; v < 8; ++v)
        out[(size_t)(trow + v)*OUT_DIM + ocol] = acc[mi][ni][v];
    }
  }
}

// ---------------------------------------------------------------------------
extern "C" void kernel_launch(void* const* d_in, const int* in_sizes, int n_in,
                              void* d_out, int out_size, void* d_ws, size_t ws_size,
                              hipStream_t stream) {
  (void)in_sizes; (void)n_in; (void)out_size; (void)ws_size;
  const float* x   = (const float*)d_in[0];
  const float* W   = (const float*)d_in[1];
  const float* Q   = (const float*)d_in[2];
  const float* R   = (const float*)d_in[3];
  const float* lam = (const float*)d_in[4];
  float*  out  = (float*)d_out;
  __bf16* weff = (__bf16*)d_ws;   // 2048*2048 bf16 = 8 MB scratch

  build_weff_kernel<<<dim3(OUT_DIM / BM, IN_DIM / BN), NTHREADS, 0, stream>>>(
      W, Q, R, lam, weff);
  qrlora_gemm_kernel<<<dim3(T_DIM / BM, OUT_DIM / BN), NTHREADS, 0, stream>>>(
      x, weff, out);
}
